// MultiHopGCN_59115929862450
// MI455X (gfx1250) — matrix operand order repaired and checked by reading, not compiled, not run on hardware
//
#include <hip/hip_runtime.h>
#include <hip/hip_bf16.h>

// MultiHop GCN forward for MI455X (gfx1250, wave32).
// GEMMs: v_wmma_f32_16x16x32_f16, A/B staged in LDS (b128 global loads,
// packed-b32 transpose stores, ds_load_b128 fragments, 4 WMMAs per K-panel,
// software-pipelined by the compiler). Message passing: float4 gathers +
// global_atomic_add_f32 scatters (h is 51MB, L2 = 192MB -> atomics land in L2).
// Workload is memory-bound on ~3-4GB of edge traffic (~150us @ 23.3 TB/s);
// 13 GFLOP of GEMM is noise on the WMMA units, so full-array passes are fused
// away (f16 converts folded into GEMM / bias-relu epilogues).

typedef __attribute__((ext_vector_type(16))) _Float16 v16h;
typedef __attribute__((ext_vector_type(8)))  float    v8f;

#define HID    128        // hidden width (fixed by the model)
#define KP     128        // K panel staged in LDS per iteration
#define LSTR   (KP + 8)   // LDS row stride in halves (pad: 4-bank shift/row)

// ---------------------------------------------------------------- utilities
__global__ void k_fill_f32(float* __restrict__ p, float v, int n) {
  int i = blockIdx.x * blockDim.x + threadIdx.x;
  if (i < n) p[i] = v;
}

__global__ void k_deg_count(const int* __restrict__ dst, float* __restrict__ deg, int E) {
  int i = blockIdx.x * blockDim.x + threadIdx.x;
  if (i < E) atomicAdd(&deg[dst[i]], 1.0f);
}

__global__ void k_rsqrt_inplace(float* __restrict__ p, int n) {
  int i = blockIdx.x * blockDim.x + threadIdx.x;
  if (i < n) p[i] = rsqrtf(p[i]);
}

__global__ void k_f32_to_f16(const float* __restrict__ s, _Float16* __restrict__ d, int n) {
  int i = blockIdx.x * blockDim.x + threadIdx.x;
  if (i < n) d[i] = (_Float16)s[i];
}

// ------------------------------------------------------------- WMMA GEMM
// C[M,128] = A[M,K] (f16 row-major) x B[K,128] (f16 row-major), f32 out.
// Requires M % 16 == 0 (holds: 100000 = 6250*16) and K % 128 == 0.
// Block = 256 threads = 8 waves; block owns 16 rows, wave w owns cols [16w,16w+16).
// K is processed in 128-wide panels staged in LDS:
//   As[r][k]  : A panel, row-major (16 x KP)
//   Bs[n][k]  : B panel, TRANSPOSED (128 x KP) so a column read is contiguous
// Fragment layouts per CDNA5 ISA 7.12.2 (wave32):
//   A 16x32 f16 : lane(0-15)=row, holds K {ka..ka+7, ka+16..ka+23}, ka=kk+8*(lane>>4)
//                 -> two contiguous 16B runs -> 2x ds_load_b128
//   B 32x16 f16 : lane holds col n=lane&15, K {kb..kb+7, kb+16..kb+23}, kb=kk+8*(lane>>4)
//                 -> two contiguous 16B runs  -> 2x ds_load_b128
//   C 16x16 f32 : VGPR r -> (lane<16 ? M=r : M=r+8), N=lane&15
__global__ void k_wmma_gemm_f16(const _Float16* __restrict__ A,
                                const _Float16* __restrict__ B,
                                const float* __restrict__ bias,      // nullable
                                float* __restrict__ C,
                                _Float16* __restrict__ C16,          // nullable f16 copy
                                int M, int K, int fuse_relu) {
  __shared__ _Float16 As[16 * LSTR];
  __shared__ _Float16 Bs[HID * LSTR];

  const int tid  = threadIdx.x;
  const int row0 = blockIdx.x * 16;
  const int lane = tid & 31;
  const int wave = tid >> 5;
  const int half = lane >> 4;
  const int l15  = lane & 15;
  const int ncol = wave * 16 + l15;

  const _Float16* __restrict__ arow = As + l15 * LSTR;
  const _Float16* __restrict__ bcol = Bs + ncol * LSTR;

  v8f acc = {};
  for (int k0 = 0; k0 < K; k0 += KP) {
    // stage A panel: 16 x 128 halves = 256 b128 chunks (1 per thread), coalesced
    {
      const int e = tid * 8;                 // tid < 256 covers all 2048 halves
      const int r = e >> 7, k = e & (KP - 1);
      uint4 d = *(const uint4*)(A + (size_t)(row0 + r) * K + k0 + k);
      *(uint4*)(As + r * LSTR + k) = d;
    }
    // stage B panel transposed. Subtile = 2 rows (k,k+1) x 8 cols: read two
    // coalesced b128 rows, write 8 packed b32 (consecutive K pairs per column).
#pragma unroll
    for (int c = 0; c < 4; ++c) {
      const int sub = tid + c * 256;         // 1024 subtiles per panel
      const int k   = (sub >> 4) * 2;        // 64 row-pairs
      const int n0  = (sub & 15) * 8;        // 16 col-groups of 8
      uint4 d0 = *(const uint4*)(B + (size_t)(k0 + k) * HID + n0);
      uint4 d1 = *(const uint4*)(B + (size_t)(k0 + k + 1) * HID + n0);
      const _Float16* r0 = (const _Float16*)&d0;
      const _Float16* r1 = (const _Float16*)&d1;
#pragma unroll
      for (int j = 0; j < 8; ++j) {
        union { _Float16 h[2]; unsigned u; } pk;
        pk.h[0] = r0[j];
        pk.h[1] = r1[j];
        *(unsigned*)(Bs + (n0 + j) * LSTR + k) = pk.u;   // 4B aligned: 272B rows, k even
      }
    }
    __syncthreads();

#pragma unroll
    for (int kk = 0; kk < KP; kk += 32) {
      union { v16h v; uint4 u[2]; } a, b;
      const int ka = kk + half * 8;
      a.u[0] = *(const uint4*)(arow + ka);
      a.u[1] = *(const uint4*)(arow + ka + 16);
      const int kb = kk + half * 8;
      b.u[0] = *(const uint4*)(bcol + kb);
      b.u[1] = *(const uint4*)(bcol + kb + 16);
      acc = __builtin_amdgcn_wmma_f32_16x16x32_f16(
          /*neg_a=*/false, a.v, /*neg_b=*/false, b.v,
          /*c_mod=*/(short)0, acc, /*reuse_a=*/false, /*reuse_b=*/false);
    }
    __syncthreads();
  }

  const float bv = bias ? bias[ncol] : 0.0f;
  const int mbase = row0 + half * 8;
#pragma unroll
  for (int r = 0; r < 8; ++r) {
    float v = acc[r] + bv;
    if (fuse_relu) v = fmaxf(v, 0.0f);
    const size_t idx = (size_t)(mbase + r) * HID + ncol;
    C[idx] = v;
    if (C16) C16[idx] = (_Float16)v;
  }
}

// -------------------------------------------------- GCN message passing
// h[i,:] = t[i,:] * dinv[i]^2   (self-loop term; full overwrite -> no memset)
__global__ void k_self_init(const float* __restrict__ t, const float* __restrict__ dinv,
                            float* __restrict__ h, int n) {
  int i = blockIdx.x * blockDim.x + threadIdx.x;     // one float4 per thread
  const int total = n * (HID / 4);
  if (i >= total) return;
  const int node = i >> 5;                            // 128/4 = 32 float4 per node
  const float d = dinv[node];
  const float c = d * d;
  float4 v = ((const float4*)t)[i];
  v.x *= c; v.y *= c; v.z *= c; v.w *= c;
  ((float4*)h)[i] = v;
}

// h[dst,:] += t[src,:] * dinv[src]*dinv[dst]; 32 lanes share one edge (float4 each)
__global__ void k_edge_scatter(const float* __restrict__ t, const float* __restrict__ dinv,
                               const int* __restrict__ src, const int* __restrict__ dst,
                               float* __restrict__ h, int E) {
  int gid = blockIdx.x * blockDim.x + threadIdx.x;
  if (gid >= E * 32) return;
  const int e  = gid >> 5;
  const int f4 = gid & 31;
  const int s = src[e];
  const int d = dst[e];
  const float c = dinv[s] * dinv[d];
  float4 v = ((const float4*)(t + (size_t)s * HID))[f4];
  float* hp = h + (size_t)d * HID + f4 * 4;
  atomicAdd(hp + 0, v.x * c);
  atomicAdd(hp + 1, v.y * c);
  atomicAdd(hp + 2, v.z * c);
  atomicAdd(hp + 3, v.w * c);
}

// h = relu(h + b), plus fused f16 copy for the next GEMM's A operand
__global__ void k_bias_relu_cvt(float* __restrict__ h, const float* __restrict__ b,
                                _Float16* __restrict__ h16, int n) {
  int i = blockIdx.x * blockDim.x + threadIdx.x;
  if (i < n * HID) {
    const int f = i & (HID - 1);
    const float v = fmaxf(h[i] + b[f], 0.0f);
    h[i] = v;
    h16[i] = (_Float16)v;
  }
}

// out[i] = b_out + sum_f h0[i,f]*W[f] + h1[i,f]*W[128+f] + h2[i,f]*W[256+f]
// one wave per node, butterfly reduce (wave32)
__global__ void k_readout(const float* __restrict__ h0, const float* __restrict__ h1,
                          const float* __restrict__ h2, const float* __restrict__ Wout,
                          const float* __restrict__ bout, float* __restrict__ out, int n) {
  const int wave = threadIdx.x >> 5;
  const int lane = threadIdx.x & 31;
  const int node = blockIdx.x * 8 + wave;
  if (node >= n) return;
  const float* r0 = h0 + (size_t)node * HID;
  const float* r1 = h1 + (size_t)node * HID;
  const float* r2 = h2 + (size_t)node * HID;
  float s = 0.0f;
#pragma unroll
  for (int f = lane; f < HID; f += 32)
    s += r0[f] * Wout[f] + r1[f] * Wout[HID + f] + r2[f] * Wout[2 * HID + f];
#pragma unroll
  for (int m = 16; m > 0; m >>= 1) s += __shfl_xor(s, m, 32);
  if (lane == 0) out[node] = s + bout[0];
}

// ----------------------------------------------------------------- driver
extern "C" void kernel_launch(void* const* d_in, const int* in_sizes, int n_in,
                              void* d_out, int out_size, void* d_ws, size_t ws_size,
                              hipStream_t stream) {
  const float* x    = (const float*)d_in[0];
  const int*   ei   = (const int*)  d_in[1];   // int32 on device (JAX x64 off)
  const float* Win  = (const float*)d_in[2];
  const float* bin  = (const float*)d_in[3];
  const float* W0   = (const float*)d_in[4];
  const float* b0   = (const float*)d_in[5];
  const float* W1   = (const float*)d_in[6];
  const float* b1   = (const float*)d_in[7];
  const float* Wout = (const float*)d_in[8];
  const float* bout = (const float*)d_in[9];
  float* out = (float*)d_out;

  const int IN = 256;
  const int N  = in_sizes[0] / IN;   // 100000 (divisible by 16)
  const int E  = in_sizes[1] / 2;    // 1600000
  const int* src = ei;
  const int* dst = ei + E;

  // workspace carve-out (256B aligned)
  char* ws = (char*)d_ws;
  size_t off = 0;
  auto take = [&](size_t bytes) -> char* {
    char* p = ws + off;
    off += (bytes + 255) & ~(size_t)255;
    return p;
  };
  _Float16* X16   = (_Float16*)take((size_t)N * IN * sizeof(_Float16));   // x as f16
  _Float16* H16   = (_Float16*)take((size_t)N * HID * sizeof(_Float16));  // cur features f16
  _Float16* Wi16  = (_Float16*)take((size_t)IN * HID * sizeof(_Float16));
  _Float16* W0_16 = (_Float16*)take((size_t)HID * HID * sizeof(_Float16));
  _Float16* W1_16 = (_Float16*)take((size_t)HID * HID * sizeof(_Float16));
  float* dinv = (float*)take((size_t)N * sizeof(float));
  float* h0   = (float*)take((size_t)N * HID * sizeof(float));
  float* h1   = (float*)take((size_t)N * HID * sizeof(float));
  float* h2   = (float*)take((size_t)N * HID * sizeof(float));
  float* t    = (float*)take((size_t)N * HID * sizeof(float));
  (void)ws_size; (void)n_in; (void)out_size;

  const int B = 256;
  auto g = [](long long n, int b) { return (unsigned)((n + b - 1) / b); };

  // 1) symmetric normalization: deg (incl. self-loop) -> dinv = rsqrt(deg)
  k_fill_f32     <<<g(N, B), B, 0, stream>>>(dinv, 1.0f, N);
  k_deg_count    <<<g(E, B), B, 0, stream>>>(dst, dinv, E);
  k_rsqrt_inplace<<<g(N, B), B, 0, stream>>>(dinv, N);

  // 2) one-time f16 conversions (weights tiny; x once)
  k_f32_to_f16<<<g(IN * HID, B), B, 0, stream>>>(Win, Wi16, IN * HID);
  k_f32_to_f16<<<g(HID * HID, B), B, 0, stream>>>(W0, W0_16, HID * HID);
  k_f32_to_f16<<<g(HID * HID, B), B, 0, stream>>>(W1, W1_16, HID * HID);
  k_f32_to_f16<<<g((long long)N * IN, B), B, 0, stream>>>(x, X16, N * IN);

  // 3) layer-in: h0 = relu(x @ W_in + b_in), f16 copy fused into epilogue
  k_wmma_gemm_f16<<<N / 16, B, 0, stream>>>(X16, Wi16, bin, h0, H16, N, IN, 1);

  // 4) GCN conv 0: t = h0 @ W0; h1 = scatter(norm * t) + b0; relu (+f16 copy)
  k_wmma_gemm_f16<<<N / 16, B, 0, stream>>>(H16, W0_16, nullptr, t, nullptr, N, HID, 0);
  k_self_init     <<<g((long long)N * 32, B), B, 0, stream>>>(t, dinv, h1, N);
  k_edge_scatter  <<<g((long long)E * 32, B), B, 0, stream>>>(t, dinv, src, dst, h1, E);
  k_bias_relu_cvt <<<g((long long)N * HID, B), B, 0, stream>>>(h1, b0, H16, N);

  // 5) GCN conv 1: same with h1 -> h2
  k_wmma_gemm_f16<<<N / 16, B, 0, stream>>>(H16, W1_16, nullptr, t, nullptr, N, HID, 0);
  k_self_init     <<<g((long long)N * 32, B), B, 0, stream>>>(t, dinv, h2, N);
  k_edge_scatter  <<<g((long long)E * 32, B), B, 0, stream>>>(t, dinv, src, dst, h2, E);
  k_bias_relu_cvt <<<g((long long)N * HID, B), B, 0, stream>>>(h2, b1, H16, N);

  // 6) readout: out = concat(h0,h1,h2) @ W_out + b_out
  k_readout<<<g(N, 8), B, 0, stream>>>(h0, h1, h2, Wout, bout, out, N);
}
